// cait_models_17746804867806
// MI455X (gfx1250) — compile-verified, matrix operand order checked by reading, not run on hardware
//
#include <hip/hip_runtime.h>
#include <math.h>

// ---------------------------------------------------------------------------
// CaiT class-attention x2 for MI455X (gfx1250, wave32, WMMA).
// GEMMs: v_wmma_f32_16x16x32_bf16, one wave computes a 16x(16*NT) strip so the
// A fragment is loaded once per k-step (8x less HBM traffic on the big
// token-projection GEMMs). Attention: flash-style streaming softmax with K/V
// tiles staged in LDS and shared by the 5 q-tile waves of each (b,h).
// ---------------------------------------------------------------------------

typedef __bf16 bf16_t;
typedef __attribute__((ext_vector_type(16))) __bf16 v16bf;
typedef __attribute__((ext_vector_type(8)))  __bf16 bf16x8;
typedef __attribute__((ext_vector_type(8)))  float  v8f;

#define CB   64      // batch
#define KTOK 75      // class tokens
#define NPAT 2025    // patch tokens
#define CC   256     // channels
#define HH   8       // heads
#define HDIM 32      // head dim
#define HID  1024
#define TTOK 2100    // K+N
#define TPAD 2112    // padded tokens (66 * 32)
#define QPAD 80      // padded cls rows (5 * 16)

static __device__ __forceinline__ bf16_t f2bf(float f) {
  unsigned u = __builtin_bit_cast(unsigned, f);
  unsigned r = (u + 0x7FFFu + ((u >> 16) & 1u)) >> 16;      // RNE
  return __builtin_bit_cast(bf16_t, (unsigned short)r);
}

// Load one 16-bit A/B WMMA fragment: elements 0..7 at p, 8..15 at p+16
// (CDNA5 16-bit A/B layout: lane holds K = {0..7,16..23} or {8..15,24..31}
// depending on lane half — caller bakes hi*8 into p).
static __device__ __forceinline__ v16bf ldfrag(const bf16_t* __restrict__ p) {
  bf16x8 a = *(const bf16x8*)p;
  bf16x8 b = *(const bf16x8*)(p + 16);
  v16bf r;
#pragma unroll
  for (int i = 0; i < 8; ++i) { r[i] = a[i]; r[8 + i] = b[i]; }
  return r;
}

// ---------------------------------------------------------------------------
// fp32 -> bf16 weight conversion
// ---------------------------------------------------------------------------
__global__ void conv_f2bf(const float* __restrict__ in, bf16_t* __restrict__ out, int n) {
  int i = blockIdx.x * blockDim.x + threadIdx.x;
  if (i < n) out[i] = f2bf(in[i]);
}

// ---------------------------------------------------------------------------
// LayerNorm over rows of 256, fp32 in -> bf16 out.  One wave per row.
// For LN1 the logical input row t is concat(x_cls, x_patch).
// ---------------------------------------------------------------------------
__global__ void __launch_bounds__(256) ln_rows(
    const float* __restrict__ xc, const float* __restrict__ xpatch,
    int Krows, int rows_per_batch, long long total_rows,
    const float* __restrict__ w, const float* __restrict__ bb,
    bf16_t* __restrict__ out, long long out_stride_b)
{
  int warp = threadIdx.x >> 5, lane = threadIdx.x & 31;
  long long row = (long long)blockIdx.x * 8 + warp;
  if (row >= total_rows) return;
  int b = (int)(row / rows_per_batch);
  int t = (int)(row % rows_per_batch);
  const float* src = (t < Krows)
      ? xc + ((long long)b * Krows + t) * CC
      : xpatch + ((long long)b * NPAT + (t - Krows)) * CC;
  float vals[8], s = 0.f, s2 = 0.f;
#pragma unroll
  for (int j = 0; j < 8; ++j) {
    float v = src[lane + 32 * j];
    vals[j] = v; s += v; s2 += v * v;
  }
#pragma unroll
  for (int off = 16; off > 0; off >>= 1) {
    s  += __shfl_xor(s,  off, 32);
    s2 += __shfl_xor(s2, off, 32);
  }
  float mean = s * (1.f / CC);
  float var  = s2 * (1.f / CC) - mean * mean;
  float inv  = rsqrtf(var + 1e-5f);
  bf16_t* op = out + (long long)b * out_stride_b + (long long)t * CC;
#pragma unroll
  for (int j = 0; j < 8; ++j) {
    int c = lane + 32 * j;
    op[c] = f2bf((vals[j] - mean) * inv * w[c] + bb[c]);
  }
}

// ---------------------------------------------------------------------------
// bf16 WMMA GEMM: D[M,N] = A[M,K] @ Bw[N,K]^T.
// One wave computes a 16 x (16*NT) strip: A fragment loaded once per k-step,
// NT weight fragments + NT WMMAs (weights are L2-resident, A is the big one).
// epil: 0 = bf16 store (*alpha)             (Q / K projections)
//       1 = bf16 transposed store           (V^T : out[(b*nchan+col)*ldc+row])
//       2 = fp32: resid + gamma*(acc+bias)  (proj / fc2 residual)
//       3 = bf16: exact GELU(acc+bias)      (fc1)
// ---------------------------------------------------------------------------
template <int NT>
__global__ void __launch_bounds__(32) gemm_wmma_bf16(
    const bf16_t* __restrict__ A, int lda, long long strideA, int Mreal, int Kdim,
    const bf16_t* __restrict__ Bw, int ldb,
    int epil, float alpha,
    const float* __restrict__ bias, const float* __restrict__ gamma,
    const float* __restrict__ resid, long long strideR,
    float* __restrict__ outF, bf16_t* __restrict__ outB,
    int ldc, long long strideO, int nchan)
{
  const int lane = threadIdx.x & 31;
  const int l15 = lane & 15, hi = lane >> 4;
  const int mt = blockIdx.x, ntb = blockIdx.y * NT, b = blockIdx.z;

  int arow = mt * 16 + l15; if (arow >= Mreal) arow = Mreal - 1;   // clamp loads

  const bf16_t* Ap = A  + (long long)b * strideA + (long long)arow * lda + hi * 8;
  const bf16_t* Bp = Bw + (long long)(ntb * 16 + l15) * ldb + hi * 8;

  v8f acc[NT];
#pragma unroll
  for (int t = 0; t < NT; ++t)
#pragma unroll
    for (int i = 0; i < 8; ++i) acc[t][i] = 0.f;

  for (int k = 0; k < Kdim; k += 32) {
    if (k + 64 < Kdim) __builtin_prefetch(Ap + k + 64, 0, 3);   // near-cache prefetch
    v16bf af = ldfrag(Ap + k);
#pragma unroll
    for (int t = 0; t < NT; ++t) {
      v16bf bf = ldfrag(Bp + (long long)t * 16 * ldb + k);
      acc[t] = __builtin_amdgcn_wmma_f32_16x16x32_bf16(
          false, af, false, bf, (short)0, acc[t], false, false);
    }
  }

#pragma unroll
  for (int t = 0; t < NT; ++t) {
    const int col = (ntb + t) * 16 + l15;
#pragma unroll
    for (int v = 0; v < 8; ++v) {
      int row = mt * 16 + v + 8 * hi;
      if (row >= Mreal) continue;
      float val = acc[t][v];
      if (epil == 0) {
        outB[(long long)b * strideO + (long long)row * ldc + col] = f2bf(val * alpha);
      } else if (epil == 1) {
        outB[((long long)b * nchan + col) * ldc + row] = f2bf(val);
      } else if (epil == 2) {
        long long oi = (long long)b * strideO + (long long)row * ldc + col;
        long long ri = (long long)b * strideR + (long long)row * ldc + col;
        outF[oi] = resid[ri] + gamma[col] * (val + bias[col]);
      } else {
        float x = val + bias[col];
        float g = 0.5f * x * (1.0f + erff(x * 0.70710678118654752f));  // exact GELU
        outB[(long long)b * strideO + (long long)row * ldc + col] = f2bf(g);
      }
    }
  }
}

// ---------------------------------------------------------------------------
// Flash class-attention. One workgroup (5 waves, 160 threads) per (head,batch);
// wave w owns q-tile w. Per 32-token iteration the K tile (32n x 32hd) and
// V^T tile (32hd x 32n) are staged into LDS once and shared by all 5 waves.
// S^T = Ktile(16n x 32hd) x Q^T(32hd x 16q)  -> lane = q column, n in regs.
// O^T = V^T(16hd x 32n) x P^T(32n x 16q)     -> P^T frag is the exp'd S^T regs
// repacked per-lane to bf16 pairs (layouts line up exactly, no shuffles).
// ---------------------------------------------------------------------------
__global__ void __launch_bounds__(160) attn_flash(
    const bf16_t* __restrict__ q,    // [B, QPAD, 256] bf16, pre-scaled by hd^-0.5
    const bf16_t* __restrict__ kk,   // [B, TPAD, 256] bf16
    const bf16_t* __restrict__ vvT,  // [B, 256, TPAD] bf16 (channel-major)
    bf16_t* __restrict__ o)          // [B, QPAD, 256] bf16
{
  __shared__ bf16_t kt[32 * 32];   // [n_local][hd]
  __shared__ bf16_t vt[32 * 32];   // [hd][n_local]

  const int tid = threadIdx.x;
  const int qt = tid >> 5, lane = tid & 31;
  const int l15 = lane & 15, hi = lane >> 4;
  const int h = blockIdx.x, b = blockIdx.y;

  // Q^T B-fragment: lane = q column, K(hd) = hi*8 + {0..7,16..23}
  const bf16_t* qp = q + ((long long)b * QPAD + qt * 16 + l15) * CC + h * HDIM + hi * 8;
  v16bf qf = ldfrag(qp);

  const bf16_t* kbase = kk  + (long long)b * TPAD * CC + h * HDIM;
  const bf16_t* vbase = vvT + ((long long)b * CC + h * HDIM) * TPAD;

  v8f accO0, accO1;
#pragma unroll
  for (int i = 0; i < 8; ++i) { accO0[i] = 0.f; accO1[i] = 0.f; }
  float m = -1e30f, l = 0.f;

  for (int n0 = 0; n0 < TPAD; n0 += 32) {
    // ---- cooperative stage of K / V^T tiles into LDS (16B per thread each) ----
    if (tid < 128) {
      int r = tid >> 2, c8 = (tid & 3) * 8;            // row 0..31, col {0,8,16,24}
      *(bf16x8*)(kt + r * 32 + c8) = *(const bf16x8*)(kbase + (long long)(n0 + r) * CC + c8);
      *(bf16x8*)(vt + r * 32 + c8) = *(const bf16x8*)(vbase + (long long)r * TPAD + n0 + c8);
    }
    __syncthreads();

    // ---- scores: two 16-row n-subtiles, fragments from LDS ----
    v8f z;
#pragma unroll
    for (int i = 0; i < 8; ++i) z[i] = 0.f;
    v16bf ka0 = ldfrag(kt + l15 * 32 + hi * 8);
    v16bf ka1 = ldfrag(kt + (16 + l15) * 32 + hi * 8);
    v8f s0 = __builtin_amdgcn_wmma_f32_16x16x32_bf16(false, ka0, false, qf, (short)0, z, false, false);
    v8f s1 = __builtin_amdgcn_wmma_f32_16x16x32_bf16(false, ka1, false, qf, (short)0, z, false, false);

    // ---- mask pad tokens, running max ----
    float sv0[8], sv1[8];
    float pm = m;
#pragma unroll
    for (int v = 0; v < 8; ++v) {
      int n = n0 + v + 8 * hi;
      sv0[v] = (n      < TTOK) ? s0[v] : -1e30f;
      sv1[v] = (n + 16 < TTOK) ? s1[v] : -1e30f;
      pm = fmaxf(pm, fmaxf(sv0[v], sv1[v]));
    }
    pm = fmaxf(pm, __shfl_xor(pm, 16, 32));   // combine lane-halves (same q col)
    float r = __expf(m - pm);
    m = pm;

    // ---- exp + partial sum ----
    float p0[8], p1[8], ps = 0.f;
#pragma unroll
    for (int v = 0; v < 8; ++v) {
      p0[v] = __expf(sv0[v] - m); ps += p0[v];
      p1[v] = __expf(sv1[v] - m); ps += p1[v];
    }
    ps += __shfl_xor(ps, 16, 32);
    l = l * r + ps;

    // ---- rescale accumulators ----
#pragma unroll
    for (int v = 0; v < 8; ++v) { accO0[v] *= r; accO1[v] *= r; }

    // ---- pack P^T B-fragment (K order: subtile0 regs then subtile1 regs) ----
    v16bf pf;
#pragma unroll
    for (int v = 0; v < 8; ++v) { pf[v] = f2bf(p0[v]); pf[8 + v] = f2bf(p1[v]); }

    // ---- V^T A-fragments (hd rows 0..15 / 16..31) from LDS ----
    v16bf va0 = ldfrag(vt + l15 * 32 + hi * 8);
    v16bf va1 = ldfrag(vt + (16 + l15) * 32 + hi * 8);
    accO0 = __builtin_amdgcn_wmma_f32_16x16x32_bf16(false, va0, false, pf, (short)0, accO0, false, false);
    accO1 = __builtin_amdgcn_wmma_f32_16x16x32_bf16(false, va1, false, pf, (short)0, accO1, false, false);

    __syncthreads();   // protect LDS tiles before next stage
  }

  // ---- finalize: o[b][qrow][h*32+hd] = accO / l ----
  float linv = 1.0f / l;
  bf16_t* op = o + ((long long)b * QPAD + qt * 16 + l15) * CC + h * HDIM;
#pragma unroll
  for (int v = 0; v < 8; ++v) {
    int hd = v + 8 * hi;
    op[hd]      = f2bf(accO0[v] * linv);
    op[16 + hd] = f2bf(accO1[v] * linv);
  }
}

// ---------------------------------------------------------------------------
// Host side
// ---------------------------------------------------------------------------
extern "C" void kernel_launch(void* const* d_in, const int* in_sizes, int n_in,
                              void* d_out, int out_size, void* d_ws, size_t ws_size,
                              hipStream_t stream)
{
  (void)in_sizes; (void)n_in; (void)out_size; (void)ws_size;

  const float* x_cls  = (const float*)d_in[0];
  const float* x_pat  = (const float*)d_in[1];
  const float* ln1_w  = (const float*)d_in[2];
  const float* ln1_b  = (const float*)d_in[3];
  const float* q_w    = (const float*)d_in[4];
  const float* k_w    = (const float*)d_in[5];
  const float* v_w    = (const float*)d_in[6];
  const float* proj_w = (const float*)d_in[7];
  const float* proj_b = (const float*)d_in[8];
  const float* ln2_w  = (const float*)d_in[9];
  const float* ln2_b  = (const float*)d_in[10];
  const float* fc1_w  = (const float*)d_in[11];
  const float* fc1_b  = (const float*)d_in[12];
  const float* fc2_w  = (const float*)d_in[13];
  const float* fc2_b  = (const float*)d_in[14];
  const float* gamma1 = (const float*)d_in[15];
  const float* gamma2 = (const float*)d_in[16];
  float* out = (float*)d_out;

  // -------- workspace layout (~237 MB) --------
  char* ws = (char*)d_ws;
  size_t off = 0;
  auto alloc = [&](size_t bytes) { size_t o = off; off = (off + bytes + 255) & ~(size_t)255; return o; };
  size_t o_wq  = alloc((size_t)2 * CC * CC * 2);
  size_t o_wk  = alloc((size_t)2 * CC * CC * 2);
  size_t o_wv  = alloc((size_t)2 * CC * CC * 2);
  size_t o_wp  = alloc((size_t)2 * CC * CC * 2);
  size_t o_wf1 = alloc((size_t)2 * HID * CC * 2);
  size_t o_wf2 = alloc((size_t)2 * CC * HID * 2);
  size_t o_un  = alloc((size_t)CB * TPAD * CC * 2);
  size_t o_kk  = alloc((size_t)CB * TPAD * CC * 2);
  size_t o_vt  = alloc((size_t)CB * CC * TPAD * 2);
  size_t o_q   = alloc((size_t)CB * QPAD * CC * 2);
  size_t o_o   = alloc((size_t)CB * QPAD * CC * 2);
  size_t o_xn  = alloc((size_t)CB * KTOK * CC * 2);
  size_t o_h   = alloc((size_t)CB * KTOK * HID * 2);
  size_t o_x1  = alloc((size_t)CB * KTOK * CC * 4);
  size_t o_x2  = alloc((size_t)CB * KTOK * CC * 4);

  bf16_t* wq  = (bf16_t*)(ws + o_wq);
  bf16_t* wk  = (bf16_t*)(ws + o_wk);
  bf16_t* wv  = (bf16_t*)(ws + o_wv);
  bf16_t* wp  = (bf16_t*)(ws + o_wp);
  bf16_t* wf1 = (bf16_t*)(ws + o_wf1);
  bf16_t* wf2 = (bf16_t*)(ws + o_wf2);
  bf16_t* un  = (bf16_t*)(ws + o_un);
  bf16_t* kkb = (bf16_t*)(ws + o_kk);
  bf16_t* vvT = (bf16_t*)(ws + o_vt);
  bf16_t* qb  = (bf16_t*)(ws + o_q);
  bf16_t* ob  = (bf16_t*)(ws + o_o);
  bf16_t* xn  = (bf16_t*)(ws + o_xn);
  bf16_t* hb  = (bf16_t*)(ws + o_h);
  float*  xcA = (float*)(ws + o_x1);
  float*  xcB = (float*)(ws + o_x2);

  // -------- convert weights fp32 -> bf16 (tiny, L2-resident) --------
  auto conv = [&](const float* src, bf16_t* dst, int n) {
    conv_f2bf<<<dim3((n + 255) / 256), dim3(256), 0, stream>>>(src, dst, n);
  };
  conv(q_w,    wq,  2 * CC * CC);
  conv(k_w,    wk,  2 * CC * CC);
  conv(v_w,    wv,  2 * CC * CC);
  conv(proj_w, wp,  2 * CC * CC);
  conv(fc1_w,  wf1, 2 * HID * CC);
  conv(fc2_w,  wf2, 2 * CC * HID);

  const float scale = 0.17677669529663687f;  // 32^-0.5

  // NT=8 strip for the big token GEMMs (A loaded once per k-step for 128 output
  // cols), NT=2 for the small M=75 GEMMs to keep enough waves in flight.
  auto gemm8 = [&](int mt, int ntil, const bf16_t* A, int lda, long long sA, int M, int Kd,
                   const bf16_t* Bw, int ldb, int epil, float alpha,
                   const float* bias, const float* gm, const float* res, long long sR,
                   float* oF, bf16_t* oB, int ldc, long long sO, int nchan) {
    gemm_wmma_bf16<8><<<dim3(mt, ntil, CB), dim3(32), 0, stream>>>(
        A, lda, sA, M, Kd, Bw, ldb, epil, alpha, bias, gm, res, sR, oF, oB, ldc, sO, nchan);
  };
  auto gemm2 = [&](int mt, int ntil, const bf16_t* A, int lda, long long sA, int M, int Kd,
                   const bf16_t* Bw, int ldb, int epil, float alpha,
                   const float* bias, const float* gm, const float* res, long long sR,
                   float* oF, bf16_t* oB, int ldc, long long sO, int nchan) {
    gemm_wmma_bf16<2><<<dim3(mt, ntil, CB), dim3(32), 0, stream>>>(
        A, lda, sA, M, Kd, Bw, ldb, epil, alpha, bias, gm, res, sR, oF, oB, ldc, sO, nchan);
  };

  const float* xc_cur = x_cls;      // fp32, [B,75,256], stride 75*256
  for (int i = 0; i < 2; ++i) {
    const long long wo  = (long long)i * CC * CC;
    const long long wo1 = (long long)i * HID * CC;

    // 1) LN1 over concat(x_cls, x_patch) -> un (bf16, TPAD-row stride)
    ln_rows<<<dim3((CB * TTOK + 7) / 8), dim3(256), 0, stream>>>(
        xc_cur, x_pat, KTOK, TTOK, (long long)CB * TTOK,
        ln1_w + i * CC, ln1_b + i * CC, un, (long long)TPAD * CC);

    // 2) Q = un[:75] @ q_w^T * scale   (bf16 out, QPAD stride)
    gemm2(5, CC / 32, un, CC, (long long)TPAD * CC, KTOK, CC,
          wq + wo, CC, 0, scale, nullptr, nullptr, nullptr, 0,
          nullptr, qb, CC, (long long)QPAD * CC, CC);

    // 3) K = un @ k_w^T  (all TPAD rows; pad rows masked later)
    gemm8(TPAD / 16, CC / 128, un, CC, (long long)TPAD * CC, TPAD, CC,
          wk + wo, CC, 0, 1.0f, nullptr, nullptr, nullptr, 0,
          nullptr, kkb, CC, (long long)TPAD * CC, CC);

    // 4) V^T = (un @ v_w^T)^T, channel-major [B,256,TPAD]
    gemm8(TPAD / 16, CC / 128, un, CC, (long long)TPAD * CC, TPAD, CC,
          wv + wo, CC, 1, 1.0f, nullptr, nullptr, nullptr, 0,
          nullptr, vvT, TPAD, 0, CC);

    // 5) flash class-attention -> ob (bf16, [B,QPAD,256]); 5 waves share LDS K/V
    attn_flash<<<dim3(HH, CB), dim3(160), 0, stream>>>(qb, kkb, vvT, ob);

    // 6) x_cls' = x_cls + gamma1 * (o @ proj_w^T + proj_b)   (fp32)
    gemm2(5, CC / 32, ob, CC, (long long)QPAD * CC, KTOK, CC,
          wp + wo, CC, 2, 1.0f, proj_b + i * CC, gamma1 + i * CC,
          xc_cur, (long long)KTOK * CC, xcA, nullptr, CC, (long long)KTOK * CC, CC);

    // 7) LN2 -> xn (bf16)
    ln_rows<<<dim3((CB * KTOK + 7) / 8), dim3(256), 0, stream>>>(
        xcA, nullptr, KTOK, KTOK, (long long)CB * KTOK,
        ln2_w + i * CC, ln2_b + i * CC, xn, (long long)KTOK * CC);

    // 8) h = GELU(xn @ fc1^T + b1) (bf16, N=1024)
    gemm8(5, HID / 128, xn, CC, (long long)KTOK * CC, KTOK, CC,
          wf1 + wo1, CC, 3, 1.0f, fc1_b + i * HID, nullptr, nullptr, 0,
          nullptr, hb, HID, (long long)KTOK * HID, HID);

    // 9) x_cls'' = x_cls' + gamma2 * (h @ fc2^T + b2)   (fp32; last writes d_out)
    float* dst = (i == 1) ? out : xcB;
    gemm2(5, CC / 32, hb, HID, (long long)KTOK * HID, KTOK, HID,
          wf2 + wo1, HID, 2, 1.0f, fc2_b + i * CC, gamma2 + i * CC,
          xcA, (long long)KTOK * CC, dst, nullptr, CC, (long long)KTOK * CC, CC);

    xc_cur = xcB;
  }
}
